// EnsemblePrompt_90529320665564
// MI455X (gfx1250) — matrix operand order repaired and checked by reading, not compiled, not run on hardware
//
#include <hip/hip_runtime.h>
#include <hip/hip_bf16.h>

#define BATCH  1024
#define FDIM   768
#define EDIM   8192
#define NACT   409
#define PSIZE  64
#define SELN   4

typedef __attribute__((ext_vector_type(16))) __bf16 v16bf;
typedef __attribute__((ext_vector_type(8)))  __bf16 v8bf;
typedef __attribute__((ext_vector_type(8)))  float  v8f;

union U16 { v16bf v; v8bf h[2]; };

__device__ __forceinline__ unsigned short f2bf(float f) {
  unsigned u = __float_as_uint(f);
  u += 0x7FFFu + ((u >> 16) & 1u);   // round-to-nearest-even
  return (unsigned short)(u >> 16);
}

// ---- kernel 1: query fp32 -> bf16 -------------------------------------------
__global__ void k_cvt_query(const float* __restrict__ q,
                            unsigned short* __restrict__ qb) {
  int i = blockIdx.x * 256 + threadIdx.x;      // grid covers BATCH*FDIM exactly
  qb[i] = f2bf(q[i]);
}

// ---- kernel 2: RP (768 x 8192) fp32 -> Bt (8192 x 768) bf16 (transpose) -----
__global__ void k_tr_rp(const float* __restrict__ rp,
                        unsigned short* __restrict__ bt) {
  __shared__ float tile[32][33];
  int n0 = blockIdx.x * 32, k0 = blockIdx.y * 32;
  for (int r = threadIdx.y; r < 32; r += 8)
    tile[r][threadIdx.x] = rp[(size_t)(k0 + r) * EDIM + n0 + threadIdx.x];
  __syncthreads();
  for (int r = threadIdx.y; r < 32; r += 8)
    bt[(size_t)(n0 + r) * FDIM + k0 + threadIdx.x] = f2bf(tile[threadIdx.x][r]);
}

// ---- kernel 3: score = query @ RP via v_wmma_f32_16x16x32_bf16 --------------
// Each wave computes a 32x32 block of C (2x2 of 16x16 tiles).
// Wave tiles: 32 (M) x 256 (N) = 8192; 8 waves/block -> 1024 blocks.
__global__ void __launch_bounds__(256) k_gemm_score(
    const unsigned short* __restrict__ qb,
    const unsigned short* __restrict__ bt,
    float* __restrict__ score) {
  const int lane  = threadIdx.x & 31;
  const int wave  = threadIdx.x >> 5;
  const int wtile = blockIdx.x * 8 + wave;
  const int nBlk  = wtile & 255;
  const int mBlk  = wtile >> 8;
  const int m0 = mBlk * 32, n0 = nBlk * 32;
  const int l16 = lane & 15;
  const int hi  = lane >> 4;                    // lane half

  v8f c00 = {}, c01 = {}, c10 = {}, c11 = {};

  // A (16-bit 16x32 layout): lane half 0 holds K 0-7 & 16-23, half 1 holds K 8-15 & 24-31
  const unsigned short* pa0 = qb + (size_t)(m0 + l16) * FDIM + hi * 8;
  const unsigned short* pa1 = pa0 + (size_t)16 * FDIM;
  // B (32x16): lane half 0 holds K 0-7 & 16-23 of column l16, half 1 holds K 8-15 & 24-31
  const unsigned short* pb0 = bt + (size_t)(n0 + l16) * FDIM + hi * 8;
  const unsigned short* pb1 = pb0 + (size_t)16 * FDIM;

  for (int k0 = 0; k0 < FDIM; k0 += 32) {
    U16 a0, a1;
    a0.h[0] = *(const v8bf*)(pa0 + k0);
    a0.h[1] = *(const v8bf*)(pa0 + k0 + 16);
    a1.h[0] = *(const v8bf*)(pa1 + k0);
    a1.h[1] = *(const v8bf*)(pa1 + k0 + 16);
    U16 ub0, ub1;
    ub0.h[0] = *(const v8bf*)(pb0 + k0);
    ub0.h[1] = *(const v8bf*)(pb0 + k0 + 16);
    ub1.h[0] = *(const v8bf*)(pb1 + k0);
    ub1.h[1] = *(const v8bf*)(pb1 + k0 + 16);
    v16bf b0 = ub0.v;
    v16bf b1 = ub1.v;
    c00 = __builtin_amdgcn_wmma_f32_16x16x32_bf16(false, a0.v, false, b0, (short)0, c00, false, false);
    c01 = __builtin_amdgcn_wmma_f32_16x16x32_bf16(false, a0.v, false, b1, (short)0, c01, false, false);
    c10 = __builtin_amdgcn_wmma_f32_16x16x32_bf16(false, a1.v, false, b0, (short)0, c10, false, false);
    c11 = __builtin_amdgcn_wmma_f32_16x16x32_bf16(false, a1.v, false, b1, (short)0, c11, false, false);
  }

  // C layout: VGPR r -> lanes 0-15: (M=r, N=lane), lanes 16-31: (M=8+r, N=lane-16)
  const int rbase = hi * 8;
  for (int r = 0; r < 8; ++r) {
    score[(size_t)(m0 + rbase + r) * EDIM + (n0 + l16)]           = c00[r];
    score[(size_t)(m0 + rbase + r) * EDIM + (n0 + 16 + l16)]      = c01[r];
    score[(size_t)(m0 + 16 + rbase + r) * EDIM + (n0 + l16)]      = c10[r];
    score[(size_t)(m0 + 16 + rbase + r) * EDIM + (n0 + 16 + l16)] = c11[r];
  }
}

// ---- kernel 4: per-row top-409 radix select + expert scores + top-4 ---------
__global__ void __launch_bounds__(256) k_select(
    const float* __restrict__ score, const float* __restrict__ mapw,
    int* __restrict__ selidx) {
  __shared__ float    s_row[EDIM];     // 32 KB
  __shared__ unsigned s_hist[256];
  __shared__ unsigned s_prefix;
  __shared__ int      s_rem;
  __shared__ int      s_cnt;
  __shared__ int      s_list[512];
  __shared__ float    s_part[256];
  __shared__ float    s_sel[64];

  const int b   = blockIdx.x;
  const int tid = threadIdx.x;
  const float4* r4 = (const float4*)(score + (size_t)b * EDIM);
  float4* s4 = (float4*)s_row;
  for (int i = tid; i < EDIM / 4; i += 256) s4[i] = r4[i];
  if (tid == 0) { s_prefix = 0u; s_rem = NACT; s_cnt = 0; }
  __syncthreads();

  // 4 x 8-bit MSB-first radix select for the NACT-th largest key
  for (int p = 0; p < 4; ++p) {
    s_hist[tid & 255] = 0;
    __syncthreads();
    const unsigned pref = s_prefix;
    const int shift = 24 - 8 * p;
    for (int i = tid; i < EDIM; i += 256) {
      unsigned u = __float_as_uint(s_row[i]);
      unsigned k = (u & 0x80000000u) ? ~u : (u | 0x80000000u);
      bool ok = (p == 0) || ((k >> (shift + 8)) == pref);
      if (ok) atomicAdd(&s_hist[(k >> shift) & 255u], 1u);
    }
    __syncthreads();
    if (tid == 0) {
      int rem = s_rem; unsigned cum = 0; int chosen = 0;
      for (int d = 255; d >= 0; --d) {
        unsigned h = s_hist[d];
        if (cum + h >= (unsigned)rem) { chosen = d; break; }
        cum += h;
      }
      s_prefix = (s_prefix << 8) | (unsigned)chosen;
      s_rem = rem - (int)cum;
    }
    __syncthreads();
  }

  // compact indices with key >= threshold (exact 409th-largest key)
  const unsigned thr = s_prefix;
  for (int i = tid; i < EDIM; i += 256) {
    unsigned u = __float_as_uint(s_row[i]);
    unsigned k = (u & 0x80000000u) ? ~u : (u | 0x80000000u);
    if (k >= thr) {
      int pos = atomicAdd(&s_cnt, 1);
      if (pos < 512) s_list[pos] = i;
    }
  }
  __syncthreads();
  const int cnt = min(s_cnt, 512);

  // expert scores: 4 groups x 64 experts; map reads are 256B coalesced
  const int g = tid >> 6, e = tid & 63;
  float acc = 0.f;
  for (int i = g; i < cnt; i += 4) {
    int idx = s_list[i];
    acc += s_row[idx] * mapw[(size_t)idx * PSIZE + e];
  }
  s_part[tid] = acc;
  __syncthreads();
  if (tid < 64)
    s_sel[tid] = s_part[tid] + s_part[64 + tid] + s_part[128 + tid] + s_part[192 + tid];
  __syncthreads();

  // top-4 of 64, descending, ties -> lowest index (matches jax top_k)
  if (tid == 0) {
    for (int s = 0; s < SELN; ++s) {
      float best = -3.4e38f; int arg = 0;
      for (int e2 = 0; e2 < PSIZE; ++e2)
        if (s_sel[e2] > best) { best = s_sel[e2]; arg = e2; }
      selidx[b * SELN + s] = arg;
      s_sel[arg] = -3.4e38f;
    }
  }
}

// ---- kernel 5: gather prompts -> out (1024 x 128 x 768) ---------------------
// out[b, p*32 + s*8 + l, :] = prompts[p, sel[b,s], l, :]
__global__ void __launch_bounds__(192) k_gather(
    const float* __restrict__ prompts, const int* __restrict__ selidx,
    float* __restrict__ out) {
  const int blk = blockIdx.x;      // b*128 + r
  const int b = blk >> 7;
  const int r = blk & 127;
  const int p = r >> 5;
  const int s = (r >> 3) & 3;
  const int l = r & 7;
  const int sel = selidx[b * SELN + s];
  const float4* src = (const float4*)(prompts +
      ((size_t)(p * PSIZE + sel) * 8 + l) * FDIM);
  float4* dst = (float4*)(out + ((size_t)b * 128 + r) * FDIM);
  dst[threadIdx.x] = src[threadIdx.x];
}

extern "C" void kernel_launch(void* const* d_in, const int* in_sizes, int n_in,
                              void* d_out, int out_size, void* d_ws, size_t ws_size,
                              hipStream_t stream) {
  const float* query   = (const float*)d_in[0];
  const float* prompts = (const float*)d_in[1];
  const float* rp      = (const float*)d_in[2];
  const float* mapw    = (const float*)d_in[3];

  char* ws = (char*)d_ws;
  float*          score  = (float*)ws;                                        // 32 MB
  unsigned short* qb     = (unsigned short*)(ws + (size_t)32 * 1024 * 1024);  // 1.5 MB
  unsigned short* btb    = (unsigned short*)(ws + (size_t)34 * 1024 * 1024);  // 12 MB
  int*            selidx = (int*)(ws + (size_t)47 * 1024 * 1024);             // 16 KB
  float* out = (float*)d_out;

  k_cvt_query<<<(BATCH * FDIM) / 256, 256, 0, stream>>>(query, qb);
  k_tr_rp<<<dim3(EDIM / 32, FDIM / 32), dim3(32, 8), 0, stream>>>(rp, btb);
  k_gemm_score<<<1024, 256, 0, stream>>>(qb, btb, score);
  k_select<<<BATCH, 256, 0, stream>>>(score, mapw, selidx);
  k_gather<<<BATCH * 128, 192, 0, stream>>>(prompts, selidx, out);
}
